// Multihead_Attention_14577119003039
// MI455X (gfx1250) — matrix operand order repaired and checked by reading, not compiled, not run on hardware
//
#include <hip/hip_runtime.h>
#include <hip/hip_bf16.h>

// Problem constants (from reference): B=4, S=2048, D=1024, H=16, HD=64
#define NB 4
#define NS 2048
#define ND 1024
#define NH 16
#define NHD 64
#define NE 3072            // 3*D
#define NM (NB * NS)       // 8192 rows for the projections
#define LDP 2052           // padded LDS row pitch in floats (breaks bank aliasing)

typedef __attribute__((ext_vector_type(16))) __bf16 bf16x16;
typedef __attribute__((ext_vector_type(8)))  __bf16 bf16x8;
typedef __attribute__((ext_vector_type(8)))  float  f32x8;
typedef __attribute__((ext_vector_type(4)))  float  f32x4;

// ---------------------------------------------------------------------------
// Fragment loaders following CDNA5 ISA 7.12.2 16-bit layouts (wave32).
// A (16x32, MxK): lane&15 = M; half=lane>>4; elems 0..7 = K[k0+half*8 ..],
//                 elems 8..15 = K[k0+16+half*8 ..]  -> two 16B loads.
// B (32x16, KxN) from NT weight rows: lane&15 = N; elems 0..7 = K[k0+half*8 ..],
//                 elems 8..15 = K[k0+16+half*8 ..]  -> two 16B loads.
// C/D (16x16 f32): VGPR r -> row = r + 8*half, col = lane&15.
// ---------------------------------------------------------------------------
__device__ __forceinline__ bf16x16 load_a_frag(const __bf16* A, int lda,
                                               int m0, int k0, int lane) {
    int m = m0 + (lane & 15);
    int half = lane >> 4;
    const __bf16* p = A + (size_t)m * lda + k0 + half * 8;
    bf16x16 r;
    ((bf16x8*)&r)[0] = *(const bf16x8*)(p);
    ((bf16x8*)&r)[1] = *(const bf16x8*)(p + 16);
    return r;
}

__device__ __forceinline__ bf16x16 load_b_frag(const __bf16* W, int ldb,
                                               int n0, int k0, int lane) {
    int n = n0 + (lane & 15);
    int half = lane >> 4;
    const __bf16* p = W + (size_t)n * ldb + k0 + half * 8;
    bf16x16 r;
    ((bf16x8*)&r)[0] = *(const bf16x8*)(p);
    ((bf16x8*)&r)[1] = *(const bf16x8*)(p + 16);
    return r;
}

__device__ __forceinline__ f32x8 wmma_bf16(bf16x16 a, bf16x16 b, f32x8 c) {
    return __builtin_amdgcn_wmma_f32_16x16x32_bf16(
        false, a, false, b, (short)0, c, false, false);
}

// ---------------------------------------------------------------------------
// f32 -> bf16 staging
// ---------------------------------------------------------------------------
__global__ void cvt_f32_bf16(const float* __restrict__ in,
                             __bf16* __restrict__ out, size_t n) {
    size_t i = (size_t)blockIdx.x * blockDim.x + threadIdx.x;
    size_t stride = (size_t)gridDim.x * blockDim.x;
    for (; i < n; i += stride) out[i] = (__bf16)in[i];
}

// ---------------------------------------------------------------------------
// Generic NT GEMM: C[M,N] = A[M,K](bf16) * W[N,K](bf16)^T * scale + bias
// Register-blocked: one wave computes 32x64 (2 M-tiles x 4 N-tiles);
// 8 independent WMMAs per 32-deep K-step hide the WMMA RAW latency.
// ---------------------------------------------------------------------------
template <bool STORE_BF16>
__global__ __launch_bounds__(128) void gemm_bf16_nt(
    const __bf16* __restrict__ A, const __bf16* __restrict__ W,
    const float* __restrict__ bias, void* __restrict__ C,
    int M, int N, int K, float scale) {
    int lane = threadIdx.x;
    int gw = blockIdx.x * 4 + threadIdx.y;
    int nblk = N >> 6;                       // 64-wide N blocks
    int mb = gw / nblk, nb = gw % nblk;
    int m0 = mb << 5, n0 = nb << 6;
    if (m0 >= M) return;

    f32x8 acc[2][4] = {};
    for (int k0 = 0; k0 < K; k0 += 32) {
        bf16x16 a0 = load_a_frag(A, K, m0, k0, lane);
        bf16x16 a1 = load_a_frag(A, K, m0 + 16, k0, lane);
#pragma unroll
        for (int j = 0; j < 4; ++j) {
            bf16x16 b = load_b_frag(W, K, n0 + 16 * j, k0, lane);
            acc[0][j] = wmma_bf16(a0, b, acc[0][j]);
            acc[1][j] = wmma_bf16(a1, b, acc[1][j]);
        }
    }
    int half = lane >> 4;
#pragma unroll
    for (int i = 0; i < 2; ++i) {
#pragma unroll
        for (int j = 0; j < 4; ++j) {
            int col = n0 + 16 * j + (lane & 15);
            float bv = bias ? bias[col] : 0.0f;
#pragma unroll
            for (int r = 0; r < 8; ++r) {
                int row = m0 + 16 * i + r + 8 * half;
                float v = acc[i][j][r] * scale + bv;
                if (STORE_BF16)
                    ((__bf16*)C)[(size_t)row * N + col] = (__bf16)v;
                else
                    ((float*)C)[(size_t)row * N + col] = v;
            }
        }
    }
}

// ---------------------------------------------------------------------------
// Transpose V into [bh, hd, S] bf16 so AV B-fragments are K-contiguous.
// ---------------------------------------------------------------------------
__global__ void transpose_v(const __bf16* __restrict__ qkv,
                            __bf16* __restrict__ vt) {
    size_t i = (size_t)blockIdx.x * blockDim.x + threadIdx.x;
    size_t stride = (size_t)gridDim.x * blockDim.x;
    size_t total = (size_t)NB * NH * NHD * NS;
    for (; i < total; i += stride) {
        int s = (int)(i % NS);
        size_t r = i / NS;
        int d = (int)(r % NHD);
        size_t r2 = r / NHD;
        int h = (int)(r2 % NH);
        int b = (int)(r2 / NH);
        vt[i] = qkv[((size_t)b * NS + s) * NE + h * (3 * NHD) + 2 * NHD + d];
    }
}

// ---------------------------------------------------------------------------
// Fused scores + softmax + AV. One workgroup (4 waves) owns a 16-query strip.
//   Phase 1: scores -> 16 x 2048 f32 strip in LDS (131 KB of the WGP's 320 KB)
//            Q fragments loaded once per wave, reused over 32 key tiles.
//   Phase 2: row softmax from LDS; probs -> LDS + nontemporal global store
//            (attention output; NT keeps the 1 GB stream out of L2).
//   Phase 3: AV; wave w computes hd-slice [16w,16w+16); probs from LDS
//            (f32->bf16 in-register), V from L2-resident vt.
// Attention tensor crosses HBM exactly once (the mandatory output write).
// qkv layout: [B,S,3D]; head h occupies e = h*192 + {q:0..63,k:64..127,v:128..191}
// ---------------------------------------------------------------------------
__global__ __launch_bounds__(128) void attn_fused(
    const __bf16* __restrict__ qkv, const __bf16* __restrict__ vt,
    float* __restrict__ attn_out, __bf16* __restrict__ values) {
    extern __shared__ float sc[];            // [16][LDP]
    int lane = threadIdx.x;
    int wid = threadIdx.y;
    int wg = blockIdx.x;                     // bh*128 + q-tile
    int bh = wg >> 7;
    int q0 = (wg & 127) << 4;
    int b = bh >> 4, h = bh & 15;
    int half = lane >> 4;
    int cl = lane & 15;

    const __bf16* qbase = qkv + (size_t)b * NS * NE + h * (3 * NHD);
    const __bf16* kbase = qbase + NHD;

    // Q fragments for kk=0 / kk=32, reused across all key tiles of this wave.
    bf16x16 aq0 = load_a_frag(qbase, NE, q0, 0, lane);
    bf16x16 aq1 = load_a_frag(qbase, NE, q0, 32, lane);

    // ---- Phase 1: scores for this wave's 512-key slice -> LDS ----
    for (int kt = 0; kt < 32; ++kt) {
        int k0 = (wid << 9) + (kt << 4);
        bf16x16 b0 = load_b_frag(kbase, NE, k0, 0, lane);
        bf16x16 b1 = load_b_frag(kbase, NE, k0, 32, lane);
        f32x8 acc = {};
        acc = wmma_bf16(aq0, b0, acc);
        acc = wmma_bf16(aq1, b1, acc);
        int col = k0 + cl;
#pragma unroll
        for (int r = 0; r < 8; ++r)
            sc[(r + 8 * half) * LDP + col] = acc[r] * 0.125f;  // 1/sqrt(64)
    }
    __syncthreads();

    // ---- Phase 2: softmax rows 4*wid .. 4*wid+3 ----
    float* gout = attn_out + (size_t)bh * NS * NS;
    for (int rr = 0; rr < 4; ++rr) {
        int row = 4 * wid + rr;
        float v[64];
        float mx = -3.402823466e38f;
#pragma unroll
        for (int i = 0; i < 64; ++i) {
            v[i] = sc[row * LDP + lane + 32 * i];
            mx = fmaxf(mx, v[i]);
        }
#pragma unroll
        for (int o = 16; o > 0; o >>= 1) mx = fmaxf(mx, __shfl_xor(mx, o, 32));
        float sum = 0.0f;
#pragma unroll
        for (int i = 0; i < 64; ++i) {
            v[i] = __expf(v[i] - mx);
            sum += v[i];
        }
#pragma unroll
        for (int o = 16; o > 0; o >>= 1) sum += __shfl_xor(sum, o, 32);
        float inv = 1.0f / sum;
#pragma unroll
        for (int i = 0; i < 64; ++i) {
            float pv = v[i] * inv;
            sc[row * LDP + lane + 32 * i] = pv;
            __builtin_nontemporal_store(
                pv, &gout[(size_t)(q0 + row) * NS + lane + 32 * i]);
        }
    }
    __syncthreads();

    // ---- Phase 3: AV; wave wid handles hd columns [16*wid, 16*wid+16) ----
    const __bf16* bbase = vt + (size_t)bh * NHD * NS;
    int n0 = wid << 4;
    f32x8 acc = {};
    for (int k0 = 0; k0 < NS; k0 += 32) {
        // A fragment (16x32) from LDS probs, cvt to bf16 (rows = lane&15)
        const float* p = &sc[cl * LDP + k0 + half * 8];
        f32x4 f0 = *(const f32x4*)(p);
        f32x4 f1 = *(const f32x4*)(p + 4);
        f32x4 f2 = *(const f32x4*)(p + 16);
        f32x4 f3 = *(const f32x4*)(p + 20);
        bf16x16 a;
#pragma unroll
        for (int i = 0; i < 4; ++i) {
            a[i]      = (__bf16)f0[i];
            a[4 + i]  = (__bf16)f1[i];
            a[8 + i]  = (__bf16)f2[i];
            a[12 + i] = (__bf16)f3[i];
        }
        bf16x16 bm = load_b_frag(bbase, NS, n0, k0, lane);
        acc = wmma_bf16(a, bm, acc);
    }
#pragma unroll
    for (int r = 0; r < 8; ++r) {
        int row = q0 + r + 8 * half;
        values[((size_t)bh * NS + row) * NHD + n0 + cl] = (__bf16)acc[r];
    }
}

// ---------------------------------------------------------------------------
extern "C" void kernel_launch(void* const* d_in, const int* in_sizes, int n_in,
                              void* d_out, int out_size, void* d_ws, size_t ws_size,
                              hipStream_t stream) {
    const float* x     = (const float*)d_in[0];   // [B,S,D]
    const float* w_qkv = (const float*)d_in[1];   // [3D,D]
    const float* b_qkv = (const float*)d_in[2];   // [3D]
    const float* w_out = (const float*)d_in[3];   // [D,D]
    const float* b_out = (const float*)d_in[4];   // [D]

    float* out  = (float*)d_out;                          // [B,S,D]
    float* attn = out + (size_t)NB * NS * ND;             // [B,H,S,S]

    // Workspace carve-up (bf16 staging buffers), ~109 MB total.
    char* ws = (char*)d_ws;
    __bf16* x_bf    = (__bf16*)ws; ws += (size_t)NM * ND * 2;
    __bf16* wqkv_bf = (__bf16*)ws; ws += (size_t)NE * ND * 2;
    __bf16* wout_bf = (__bf16*)ws; ws += (size_t)ND * ND * 2;
    __bf16* qkv_bf  = (__bf16*)ws; ws += (size_t)NM * NE * 2;
    __bf16* vt_bf   = (__bf16*)ws; ws += (size_t)NB * NH * NHD * NS * 2;
    __bf16* val_bf  = (__bf16*)ws; ws += (size_t)NM * ND * 2;

    dim3 wblk(32, 4);

    // 1) bf16 staging
    cvt_f32_bf16<<<8192, 256, 0, stream>>>(x, x_bf, (size_t)NM * ND);
    cvt_f32_bf16<<<4096, 256, 0, stream>>>(w_qkv, wqkv_bf, (size_t)NE * ND);
    cvt_f32_bf16<<<2048, 256, 0, stream>>>(w_out, wout_bf, (size_t)ND * ND);

    // 2) QKV projection: [8192,1024] x [3072,1024]^T -> bf16 qkv
    gemm_bf16_nt<true><<<(NM / 32) * (NE / 64) / 4, wblk, 0, stream>>>(
        x_bf, wqkv_bf, b_qkv, qkv_bf, NM, NE, ND, 1.0f);

    // 3) V transpose (vt is L2-resident for the fused kernel)
    transpose_v<<<8192, 256, 0, stream>>>(qkv_bf, vt_bf);

    // 4) Fused scores + softmax + AV: one workgroup per 16-query strip,
    //    131 KB dynamic LDS (CDNA5: up to 320 KB per workgroup).
    attn_fused<<<(NB * NH) * (NS / 16), wblk, 16 * LDP * sizeof(float),
                 stream>>>(qkv_bf, vt_bf, attn, val_bf);

    // 5) Output projection into out region of d_out
    gemm_bf16_nt<false><<<(NM / 32) * (ND / 64) / 4, wblk, 0, stream>>>(
        val_bf, wout_bf, b_out, out, NM, ND, ND, 1.0f);
}
